// OptimizedGAT_9680856285340
// MI455X (gfx1250) — matrix operand order repaired and checked by reading, not compiled, not run on hardware
//
#include <hip/hip_runtime.h>
#include <hip/hip_bf16.h>
#include <math.h>

typedef __attribute__((ext_vector_type(16))) _Float16 v16h;
typedef __attribute__((ext_vector_type(8)))  _Float16 v8h;
typedef __attribute__((ext_vector_type(4)))  _Float16 v4h;
typedef __attribute__((ext_vector_type(8)))  float    v8f;

// ---------------- device helpers ----------------
__device__ __forceinline__ float geluf(float x) {
  return 0.5f * x * (1.0f + erff(x * 0.70710678118654752440f));
}
__device__ __forceinline__ float lrelu(float x) { return x >= 0.0f ? x : 0.2f * x; }
__device__ __forceinline__ void atomicMaxF(float* a, float v) {
  if (v >= 0.0f) atomicMax((int*)a, __float_as_int(v));
  else           atomicMin((unsigned int*)a, __float_as_uint(v));
}

// ---------------- f32 -> f16 weight cast ----------------
__global__ void k_cast_h(const float* __restrict__ s, _Float16* __restrict__ d, int n) {
  int i = blockIdx.x * blockDim.x + threadIdx.x;
  if (i < n) d[i] = (_Float16)s[i];
}

// ---------------- fused WMMA GEMM (+bias/LN/GELU) ----------------
// C[M,Nout] = A[M,K] @ W[K,Nout].  64 rows per block, one wave per 16-col tile,
// 4 M-tiles per wave (4 independent WMMA accumulation chains).
// LDS: A as f16 [64][K+8]; W staged TRANSPOSED [Nout][K+8] so each lane's 16
// K-values are contiguous -> ds_load_b128 fragment loads.
// flags: 1=bias, 2=layernorm(gamma,beta), 4=gelu
template <int K, int Nout>
__global__ __launch_bounds__(256) void k_gemm(
    const float* __restrict__ A, const _Float16* __restrict__ W,
    const float* __restrict__ bias, const float* __restrict__ gamma,
    const float* __restrict__ beta, float* __restrict__ out,
    int M, int flags)
{
  constexpr int ROWS = 64;
  constexpr int AS = K + 8;       // padded LDS strides (halves); byte stride stays 16B-aligned
  constexpr int WS = K + 8;
  __shared__ __align__(16) _Float16 As[ROWS * AS];
  __shared__ __align__(16) _Float16 Wt[Nout * WS];
  __shared__ float Os[ROWS * Nout];
  __shared__ float st[ROWS * 2];

  const int tid  = threadIdx.x;
  const int nth  = 2 * Nout;      // blockDim
  const int row0 = blockIdx.x * ROWS;

  // A tile: float4 global loads -> 4-half LDS stores
  for (int idx = tid; idx < ROWS * (K / 4); idx += nth) {
    int r = idx / (K / 4), q = idx - r * (K / 4);
    int gr = row0 + r; if (gr >= M) gr = M - 1;
    float4 v = ((const float4*)(A + (size_t)gr * K))[q];
    *(v4h*)&As[r * AS + q * 4] =
        (v4h){(_Float16)v.x, (_Float16)v.y, (_Float16)v.z, (_Float16)v.w};
  }
  // W tile, transposed into LDS (coalesced global reads)
  for (int idx = tid; idx < K * Nout; idx += nth) {
    int k = idx / Nout, n = idx - k * Nout;
    Wt[n * WS + k] = W[idx];
  }
  __syncthreads();

  const int lane = tid & 31;
  const int n0   = (tid >> 5) << 4;
  const int half = lane >> 4;
  const int l15  = lane & 15;

  v8f c[4];
#pragma unroll
  for (int rt = 0; rt < 4; ++rt) c[rt] = (v8f){};

#pragma unroll
  for (int k0 = 0; k0 < K; k0 += 32) {
    // B fragment: lane = column, 16 contiguous K halves (2x b128)
    v8h b0 = *(const v8h*)&Wt[(n0 + l15) * WS + k0 + half * 16];
    v8h b1 = *(const v8h*)&Wt[(n0 + l15) * WS + k0 + half * 16 + 8];
    v16h b = __builtin_shufflevector(b0, b1, 0,1,2,3,4,5,6,7,8,9,10,11,12,13,14,15);
#pragma unroll
    for (int rt = 0; rt < 4; ++rt) {
      // A fragment: lane = row, 16 contiguous K halves (2x b128)
      v8h a0 = *(const v8h*)&As[(rt * 16 + l15) * AS + k0 + half * 16];
      v8h a1 = *(const v8h*)&As[(rt * 16 + l15) * AS + k0 + half * 16 + 8];
      v16h a = __builtin_shufflevector(a0, a1, 0,1,2,3,4,5,6,7,8,9,10,11,12,13,14,15);
      c[rt] = __builtin_amdgcn_wmma_f32_16x16x32_f16(false, a, false, b, (short)0, c[rt], false, false);
    }
  }
#pragma unroll
  for (int rt = 0; rt < 4; ++rt)
#pragma unroll
    for (int r = 0; r < 8; ++r)   // C layout: VGPR r -> M = r + half*8
      Os[(rt * 16 + r + half * 8) * Nout + n0 + l15] = c[rt][r];
  __syncthreads();

  if (flags & 1) {
    for (int idx = tid; idx < ROWS * Nout; idx += nth) Os[idx] += bias[idx & (Nout - 1)];
    __syncthreads();
  }
  if (flags & 2) {
    if (tid < ROWS) {
      float mu = 0.f;
      for (int n = 0; n < Nout; ++n) mu += Os[tid * Nout + n];
      mu *= (1.0f / Nout);
      float va = 0.f;
      for (int n = 0; n < Nout; ++n) { float d = Os[tid * Nout + n] - mu; va += d * d; }
      va *= (1.0f / Nout);
      st[tid * 2] = mu; st[tid * 2 + 1] = rsqrtf(va + 1e-5f);
    }
    __syncthreads();
  }
  for (int idx = tid; idx < ROWS * Nout; idx += nth) {
    int r = idx / Nout, n = idx & (Nout - 1);
    int gr = row0 + r;
    if (gr >= M) continue;
    float v = Os[idx];
    if (flags & 2) v = (v - st[r * 2]) * st[r * 2 + 1] * gamma[n] + beta[n];
    if (flags & 4) v = geluf(v);
    out[(size_t)gr * Nout + n] = v;
  }
}

// ---------------- GAT attention kernels ----------------
__global__ void k_attn(const float* __restrict__ z, const float* __restrict__ a_s,
                       const float* __restrict__ a_d, float* __restrict__ as_,
                       float* __restrict__ ad_, int Nn) {
  int i = blockIdx.x * blockDim.x + threadIdx.x;
  if (i >= Nn * 4) return;
  int n = i >> 2, h = i & 3;
  const float* zr = z + (size_t)n * 128 + h * 32;
  float s1 = 0.f, s2 = 0.f;
  for (int c = 0; c < 32; ++c) { float zv = zr[c]; s1 = fmaf(zv, a_s[h * 32 + c], s1); s2 = fmaf(zv, a_d[h * 32 + c], s2); }
  as_[i] = s1; ad_[i] = s2;
}

__global__ void k_minit(const float* __restrict__ as_, const float* __restrict__ ad_,
                        float* __restrict__ m, int cnt) {
  int i = blockIdx.x * blockDim.x + threadIdx.x;
  if (i < cnt) m[i] = lrelu(as_[i] + ad_[i]);
}

__global__ void k_emax(const int* __restrict__ src, const int* __restrict__ dst,
                       const float* __restrict__ as_, const float* __restrict__ ad_,
                       float* __restrict__ abuf, float* __restrict__ m, int E) {
  int i = blockIdx.x * blockDim.x + threadIdx.x;
  if (i >= E * 4) return;
  int e = i >> 2, h = i & 3;
  int s = src[e], d = dst[e];
  float a = lrelu(as_[s * 4 + h] + ad_[d * 4 + h]);
  abuf[i] = a;
  atomicMaxF(&m[d * 4 + h], a);
}

__global__ void k_self(const float* __restrict__ z, const float* __restrict__ as_,
                       const float* __restrict__ ad_, const float* __restrict__ m,
                       float* __restrict__ s, float* __restrict__ acc, int Nn) {
  int i = blockIdx.x * blockDim.x + threadIdx.x;
  if (i >= Nn * 128) return;
  int n = i >> 7, c = i & 127, h = c >> 5;
  float a = lrelu(as_[n * 4 + h] + ad_[n * 4 + h]);
  float e = expf(a - m[n * 4 + h]);
  acc[i] = z[i] * e;
  if ((c & 31) == 0) s[n * 4 + h] = e;
}

__global__ void k_eexp(const int* __restrict__ dst, const float* __restrict__ abuf,
                       const float* __restrict__ m, float* __restrict__ ebuf,
                       float* __restrict__ s, int E) {
  int i = blockIdx.x * blockDim.x + threadIdx.x;
  if (i >= E * 4) return;
  int e = i >> 2, h = i & 3;
  int d = dst[e];
  float ev = expf(abuf[i] - m[d * 4 + h]);
  ebuf[i] = ev;
  atomicAdd(&s[d * 4 + h], ev);
}

// edge scatter: one thread per (edge, group-of-4 channels); float4 z loads
__global__ void k_escat(const int* __restrict__ src, const int* __restrict__ dst,
                        const float* __restrict__ z, const float* __restrict__ ebuf,
                        float* __restrict__ acc, int E) {
  long long i = (long long)blockIdx.x * blockDim.x + threadIdx.x;
  if (i >= (long long)E * 32) return;
  int e = (int)(i >> 5), q = (int)(i & 31);
  int h = q >> 3;                       // q*4 channels -> head = (q*4)/32
  int sI = src[e], d = dst[e];
  float w = ebuf[e * 4 + h];
  float4 zv = ((const float4*)(z + (size_t)sI * 128))[q];
  float* ap = acc + (size_t)d * 128 + q * 4;
  atomicAdd(ap + 0, zv.x * w);
  atomicAdd(ap + 1, zv.y * w);
  atomicAdd(ap + 2, zv.z * w);
  atomicAdd(ap + 3, zv.w * w);
}

// finalize: h_out = gelu(ln(acc/s + b)) + identity  (block per node, wave32 reductions)
__global__ __launch_bounds__(128) void k_gfin(
    const float* __restrict__ acc, const float* __restrict__ s,
    const float* __restrict__ b, const float* __restrict__ g,
    const float* __restrict__ be, const float* __restrict__ ident,
    float* __restrict__ hout, int Nn) {
  __shared__ float red[4];
  int n = blockIdx.x, c = threadIdx.x;
  int h = c >> 5;
  float v = acc[(size_t)n * 128 + c] / (s[n * 4 + h] + 1e-16f) + b[c];
  float x = v;
  for (int o = 16; o > 0; o >>= 1) x += __shfl_xor(x, o, 32);
  if ((c & 31) == 0) red[c >> 5] = x;
  __syncthreads();
  float mean = (red[0] + red[1] + red[2] + red[3]) * (1.0f / 128.0f);
  float dv = v - mean;
  __syncthreads();
  float x2 = dv * dv;
  for (int o = 16; o > 0; o >>= 1) x2 += __shfl_xor(x2, o, 32);
  if ((c & 31) == 0) red[c >> 5] = x2;
  __syncthreads();
  float var = (red[0] + red[1] + red[2] + red[3]) * (1.0f / 128.0f);
  float y = geluf(dv * rsqrtf(var + 1e-5f) * g[c] + be[c]);
  hout[(size_t)n * 128 + c] = y + ident[(size_t)n * 128 + c];
}

// ---------------- pooling ----------------
__global__ void k_pool_init(float* cnt, float* xs, float* xmax) {
  int i = blockIdx.x * blockDim.x + threadIdx.x;
  if (i < 64) cnt[i] = 0.f;
  if (i < 64 * 32) { xs[i] = 0.f; xmax[i] = -3.402823466e38f; }
}
__global__ void k_pool(const float* __restrict__ hpp, const int* __restrict__ batch,
                       float* cnt, float* xs, float* xmax, int Nn) {
  int i = blockIdx.x * blockDim.x + threadIdx.x;
  if (i >= Nn * 32) return;
  int n = i >> 5, c = i & 31;
  int g = batch[n];
  float v = hpp[i];
  atomicAdd(&xs[g * 32 + c], v);
  atomicMaxF(&xmax[g * 32 + c], v);
  if (c == 0) atomicAdd(&cnt[g], 1.0f);
}
__global__ void k_poolfin(const float* cnt, const float* xs, const float* xmax,
                          const float* pool_w, float* emb) {
  int i = blockIdx.x * blockDim.x + threadIdx.x;
  if (i >= 64 * 32) return;
  int g = i >> 5, c = i & 31;
  float w0 = pool_w[0], w1 = pool_w[1], w2 = pool_w[2];
  float mx = fmaxf(w0, fmaxf(w1, w2));
  float e0 = expf(w0 - mx), e1 = expf(w1 - mx), e2 = expf(w2 - mx);
  float den = e0 + e1 + e2;
  float sv = xs[i], cn = fmaxf(cnt[g], 1.0f);
  emb[g * 96 + c]      = sv / cn * (e0 / den);
  emb[g * 96 + 32 + c] = xmax[i] * (e1 / den);
  emb[g * 96 + 64 + c] = sv * (e2 / den);
}

// ---------------- classifier (G=64 rows; one thread per graph) ----------------
__global__ __launch_bounds__(64) void k_cls(
    const float* __restrict__ emb,
    const float* W1, const float* b1, const float* g1, const float* be1,
    const float* W2, const float* b2, const float* g2, const float* be2,
    const float* W3, const float* b3, float* __restrict__ out, int G_) {
  int g = blockIdx.x * blockDim.x + threadIdx.x;
  if (g >= G_) return;
  float e[96], c1[64], c2[32];
  for (int j = 0; j < 96; ++j) e[j] = emb[g * 96 + j];
  for (int o = 0; o < 64; ++o) { float s = b1[o]; for (int j = 0; j < 96; ++j) s = fmaf(e[j], W1[j * 64 + o], s); c1[o] = s; }
  float m = 0.f; for (int o = 0; o < 64; ++o) m += c1[o]; m *= (1.0f / 64.0f);
  float v = 0.f; for (int o = 0; o < 64; ++o) { float d = c1[o] - m; v += d * d; } v *= (1.0f / 64.0f);
  float rs = rsqrtf(v + 1e-5f);
  for (int o = 0; o < 64; ++o) c1[o] = geluf((c1[o] - m) * rs * g1[o] + be1[o]);
  for (int o = 0; o < 32; ++o) { float s = b2[o]; for (int j = 0; j < 64; ++j) s = fmaf(c1[j], W2[j * 32 + o], s); c2[o] = s; }
  m = 0.f; for (int o = 0; o < 32; ++o) m += c2[o]; m *= (1.0f / 32.0f);
  v = 0.f; for (int o = 0; o < 32; ++o) { float d = c2[o] - m; v += d * d; } v *= (1.0f / 32.0f);
  rs = rsqrtf(v + 1e-5f);
  for (int o = 0; o < 32; ++o) c2[o] = geluf((c2[o] - m) * rs * g2[o] + be2[o]);
  for (int o = 0; o < 2; ++o) { float s = b3[o]; for (int j = 0; j < 32; ++j) s = fmaf(c2[j], W3[j * 2 + o], s); out[g * 2 + o] = s; }
}

// ---------------- host launch ----------------
// Input flat index map (setup_inputs insertion order, params flattened depth-first):
// 0:x 1:edge_index 2:batch 3:in_W 4:in_b 5:in_g 6:in_beta
// 7+8l..14+8l : layer l {W,b,a_s,a_d,g,beta,sW,sb}
// 39..42: pp1  43..46: pp2  47: pool_w  48..51: cl1  52..55: cl2  56,57: cl3
extern "C" void kernel_launch(void* const* d_in, const int* in_sizes, int n_in,
                              void* d_out, int out_size, void* d_ws, size_t ws_size,
                              hipStream_t stream) {
  const int N = in_sizes[0] / 128;
  const int E = in_sizes[1] / 2;
  const int G = 64;

  const float* x   = (const float*)d_in[0];
  const int* ei    = (const int*)d_in[1];
  const int* srcI  = ei;
  const int* dstI  = ei + E;
  const int* batch = (const int*)d_in[2];
  auto P = [&](int i) { return (const float*)d_in[i]; };

  // workspace carve-out
  char* ws = (char*)d_ws;
  size_t off = 0;
  auto take = [&](size_t bytes) -> char* {
    char* p = ws + off; off = (off + bytes + 255) & ~(size_t)255; return p;
  };
  _Float16* h16base = (_Float16*)take(131072 * sizeof(_Float16));
  float* big0 = (float*)take((size_t)N * 128 * 4);
  float* big1 = (float*)take((size_t)N * 128 * 4);
  float* big2 = (float*)take((size_t)N * 128 * 4);
  float* big3 = (float*)take((size_t)N * 128 * 4);
  float* h0   = (float*)take((size_t)N * 32 * 4);
  float* as_  = (float*)take((size_t)N * 4 * 4);
  float* ad_  = (float*)take((size_t)N * 4 * 4);
  float* m_   = (float*)take((size_t)N * 4 * 4);
  float* s_   = (float*)take((size_t)N * 4 * 4);
  float* abuf = (float*)take((size_t)E * 4 * 4);
  float* ebuf = (float*)take((size_t)E * 4 * 4);
  float* cnt  = (float*)take(64 * 4);
  float* xs   = (float*)take(64 * 32 * 4);
  float* xmax = (float*)take(64 * 32 * 4);
  float* emb  = (float*)take(64 * 96 * 4);
  (void)ws_size; (void)n_in; (void)out_size;

  // f16 weight cache
  size_t ho = 0;
  auto takeH = [&](size_t n) -> _Float16* { _Float16* p = h16base + ho; ho += n; return p; };
  _Float16* w_in16 = takeH(128 * 32);
  _Float16* W16[4]; _Float16* sW16[4];
  for (int l = 0; l < 4; ++l) {
    int Kl = (l == 0) ? 32 : 128;
    W16[l]  = takeH((size_t)Kl * 128);
    sW16[l] = takeH((size_t)Kl * 128);
  }
  _Float16* pp1_16 = takeH(128 * 64);
  _Float16* pp2_16 = takeH(64 * 32);

  auto cast16 = [&](const float* s, _Float16* d, int n) {
    k_cast_h<<<(n + 255) / 256, 256, 0, stream>>>(s, d, n);
  };
  cast16(P(3), w_in16, 128 * 32);
  for (int l = 0; l < 4; ++l) {
    int Kl = (l == 0) ? 32 : 128;
    cast16(P(7 + 8 * l + 0), W16[l],  Kl * 128);
    cast16(P(7 + 8 * l + 6), sW16[l], Kl * 128);
  }
  cast16(P(39), pp1_16, 128 * 64);
  cast16(P(43), pp2_16, 64 * 32);

  const int gridM = (N + 63) / 64;

  // input projection: [N,128]@[128,32] +b -> LN -> GELU
  k_gemm<128, 32><<<gridM, 64, 0, stream>>>(x, w_in16, P(4), P(5), P(6), h0, N, 7);

  const float* hin = h0;
  for (int l = 0; l < 4; ++l) {
    const int pi = 7 + 8 * l;
    float* ident = big0;
    float* z     = big1;
    float* acc   = (l & 1) ? big3 : big2;

    if (l == 0) {
      k_gemm<32, 128><<<gridM, 256, 0, stream>>>(hin, sW16[l], P(pi + 7), nullptr, nullptr, ident, N, 1);
      k_gemm<32, 128><<<gridM, 256, 0, stream>>>(hin, W16[l],  nullptr,   nullptr, nullptr, z,     N, 0);
    } else {
      k_gemm<128, 128><<<gridM, 256, 0, stream>>>(hin, sW16[l], P(pi + 7), nullptr, nullptr, ident, N, 1);
      k_gemm<128, 128><<<gridM, 256, 0, stream>>>(hin, W16[l],  nullptr,   nullptr, nullptr, z,     N, 0);
    }

    k_attn <<<(N * 4 + 255) / 256, 256, 0, stream>>>(z, P(pi + 2), P(pi + 3), as_, ad_, N);
    k_minit<<<(N * 4 + 255) / 256, 256, 0, stream>>>(as_, ad_, m_, N * 4);
    k_emax <<<(E * 4 + 255) / 256, 256, 0, stream>>>(srcI, dstI, as_, ad_, abuf, m_, E);
    k_self <<<(N * 128 + 255) / 256, 256, 0, stream>>>(z, as_, ad_, m_, s_, acc, N);
    k_eexp <<<(E * 4 + 255) / 256, 256, 0, stream>>>(dstI, abuf, m_, ebuf, s_, E);
    long long sc = (long long)E * 32;
    k_escat<<<(unsigned)((sc + 255) / 256), 256, 0, stream>>>(srcI, dstI, z, ebuf, acc, E);
    k_gfin <<<N, 128, 0, stream>>>(acc, s_, P(pi + 1), P(pi + 4), P(pi + 5), ident, acc, N);

    hin = acc;
  }

  // post-process MLP
  k_gemm<128, 64><<<gridM, 128, 0, stream>>>(hin, pp1_16, P(40), P(41), P(42), big0, N, 7);
  k_gemm<64, 32><<<gridM, 64, 0, stream>>>(big0, pp2_16, P(44), P(45), P(46), h0, N, 3);

  // pooling
  k_pool_init<<<(64 * 32 + 255) / 256, 256, 0, stream>>>(cnt, xs, xmax);
  k_pool<<<(N * 32 + 255) / 256, 256, 0, stream>>>(h0, batch, cnt, xs, xmax, N);
  k_poolfin<<<(64 * 32 + 255) / 256, 256, 0, stream>>>(cnt, xs, xmax, P(47), emb);

  // classifier
  k_cls<<<1, 64, 0, stream>>>(emb, P(48), P(49), P(50), P(51),
                              P(52), P(53), P(54), P(55), P(56), P(57),
                              (float*)d_out, G);
}
